// MeshGraphNet_66649302499639
// MI455X (gfx1250) — matrix operand order repaired and checked by reading, not compiled, not run on hardware
//
#include <hip/hip_runtime.h>

typedef __attribute__((ext_vector_type(16))) __bf16 v16bf;
typedef __attribute__((ext_vector_type(8)))  __bf16 v8bf;
typedef __attribute__((ext_vector_type(8)))  float  v8f;

#define HDIM   128
#define NNODES 50000
#define NEDGES 800000
#define LNEPS  1e-5f

// ---------------- device helpers ----------------

__device__ inline v8f v8f_zero() {
  v8f z;
#pragma unroll
  for (int i = 0; i < 8; ++i) z[i] = 0.f;
  return z;
}

// D = A*B + C  (bf16 in, f32 acc), 16x16x32
__device__ inline v8f wmma_bf16(v16bf a, v16bf b, v8f c) {
  return __builtin_amdgcn_wmma_f32_16x16x32_bf16(false, a, false, b, (short)0, c,
                                                 false, false);
}

// A-operand fragment from a row-major bf16 row (ISA 16-bit A layout):
// lanes 0-15: K = kc+[0..7], kc+16+[0..7]; lanes 16-31: +8 on each group.
__device__ inline v16bf load_arow(const __bf16* rowptr, int kc, int khalf) {
  const int kb = khalf * 8;
  union { v16bf v; v8bf h[2]; } u;
  u.h[0] = *(const v8bf*)(rowptr + kc + kb);
  u.h[1] = *(const v8bf*)(rowptr + kc + kb + 16);
  return u.v;
}

// B-operand fragment from pre-packed weights: packed[(t*32+lane)*16 + e]
__device__ inline v16bf load_btile(const __bf16* wp, int t, int lane) {
  const __bf16* p = wp + (((size_t)t * 32 + lane) << 4);
  union { v16bf v; v8bf h[2]; } u;
  u.h[0] = *(const v8bf*)p;
  u.h[1] = *(const v8bf*)(p + 8);
  return u.v;
}

// A fragment built from a short fp32 row, zero-padded to K=32 (encoders)
__device__ inline v16bf build_a_f32(const float* rowptr, int Kact, int khalf) {
  union { v16bf v; __bf16 e[16]; } u;
  const int kb = khalf * 8;
#pragma unroll
  for (int i = 0; i < 16; ++i) {
    int k = (i < 8) ? (kb + i) : (16 + kb + (i - 8));
    float f = (k < Kact) ? rowptr[k] : 0.f;
    u.e[i] = (__bf16)f;
  }
  return u.v;
}

// A fragment from fp32 row scaled by s (agg/deg), converted to bf16
__device__ inline v16bf build_a_f32scale(const float* rowptr, int kc, float s,
                                         int khalf) {
  union { v16bf v; __bf16 e[16]; } u;
  const int kb = khalf * 8;
#pragma unroll
  for (int i = 0; i < 16; ++i) {
    int k = kc + ((i < 8) ? (kb + i) : (16 + kb + (i - 8)));
    u.e[i] = (__bf16)(rowptr[k] * s);
  }
  return u.v;
}

// sum across the 16 lanes of a half-wave (C-layout rows live per half)
__device__ inline float hsum16(float x) {
  x += __shfl_xor(x, 1, 32);
  x += __shfl_xor(x, 2, 32);
  x += __shfl_xor(x, 4, 32);
  x += __shfl_xor(x, 8, 32);
  return x;
}

// ---------------- aux kernels ----------------

__global__ void zero_f32(float* __restrict__ p, long n) {
  long i = (long)blockIdx.x * blockDim.x + threadIdx.x;
  if (i < n) p[i] = 0.f;
}

__global__ void deg_kernel(const int* __restrict__ ei, float* __restrict__ deg) {
  long i = (long)blockIdx.x * blockDim.x + threadIdx.x;
  if (i < NEDGES) atomicAdd(&deg[ei[NEDGES + i]], 1.f);
}

__global__ void invdeg_kernel(float* __restrict__ deg) {
  long i = (long)blockIdx.x * blockDim.x + threadIdx.x;
  if (i < NNODES) deg[i] = 1.f / fmaxf(deg[i], 1.f);
}

// Repack fp32 [K x N] weight into bf16 WMMA B-operand tiles, zero padded.
// packed[((kt*NTn+nt)*32 + lane)*16 + e], k = kt*32 + (lane>>4)*8 + (e&7) + (e>>3)*16,
// n = nt*16 + (lane&15).
__global__ void repack_w(const float* __restrict__ W, __bf16* __restrict__ out,
                         int Kact, int Nact, int NTn, int total) {
  int idx = blockIdx.x * blockDim.x + threadIdx.x;
  if (idx >= total) return;
  int e = idx & 15, lane = (idx >> 4) & 31, t = idx >> 9;
  int nt = t % NTn, kt = t / NTn;
  int k = kt * 32 + (lane >> 4) * 8 + (e & 7) + (e >> 3) * 16;
  int n = nt * 16 + (lane & 15);
  float v = (k < Kact && n < Nact) ? W[(size_t)k * Nact + n] : 0.f;
  out[idx] = (__bf16)v;
}

// ---------------- encoder: Linear(Kin->128)+ReLU+Linear(128->128) ----------------

__global__ __launch_bounds__(256) void encoder_kernel(
    const float* __restrict__ in, int Kin, long M,
    const __bf16* __restrict__ W1p, const float* __restrict__ b1,
    const __bf16* __restrict__ W2p, const float* __restrict__ b2,
    float* __restrict__ out_f32, __bf16* __restrict__ out_bf) {
  __shared__ __bf16 lds_mid[8][16 * HDIM];
  const int lane = threadIdx.x & 31;
  const int w = threadIdx.x >> 5;
  const int col = lane & 15;
  const int khalf = lane >> 4;
  const int rowoff = khalf * 8;
  const long tbase = ((long)blockIdx.x * 8 + w) * 16;

  long ar = tbase + col;
  if (ar >= M) ar = M - 1;

  v8f acc[8];
#pragma unroll
  for (int nt = 0; nt < 8; ++nt) acc[nt] = v8f_zero();

  // layer 1: single padded K=32 chunk
  v16bf a0 = build_a_f32(in + (size_t)ar * Kin, Kin, khalf);
#pragma unroll
  for (int nt = 0; nt < 8; ++nt)
    acc[nt] = wmma_bf16(a0, load_btile(W1p, nt, lane), acc[nt]);

#pragma unroll
  for (int nt = 0; nt < 8; ++nt) {
    float bb = b1[nt * 16 + col];
#pragma unroll
    for (int m = 0; m < 8; ++m) {
      float v = acc[nt][m] + bb;
      acc[nt][m] = v > 0.f ? v : 0.f;
    }
  }

#pragma unroll
  for (int m = 0; m < 8; ++m)
#pragma unroll
    for (int nt = 0; nt < 8; ++nt)
      lds_mid[w][(rowoff + m) * HDIM + nt * 16 + col] = (__bf16)acc[nt][m];
  asm volatile("s_wait_dscnt 0" ::: "memory");

#pragma unroll
  for (int nt = 0; nt < 8; ++nt) acc[nt] = v8f_zero();
  const __bf16* midrow = &lds_mid[w][col * HDIM];
#pragma unroll
  for (int c = 0; c < 4; ++c) {
    v16bf a = load_arow(midrow, c * 32, khalf);
#pragma unroll
    for (int nt = 0; nt < 8; ++nt)
      acc[nt] = wmma_bf16(a, load_btile(W2p, c * 8 + nt, lane), acc[nt]);
  }

#pragma unroll
  for (int nt = 0; nt < 8; ++nt) {
    float bb = b2[nt * 16 + col];
#pragma unroll
    for (int m = 0; m < 8; ++m) {
      long row = tbase + rowoff + m;
      if (row < M) {
        float val = acc[nt][m] + bb;
        size_t off = (size_t)row * HDIM + nt * 16 + col;
        out_f32[off] = val;
        out_bf[off] = (__bf16)val;
      }
    }
  }
}

// ---------------- edge block: concat(h[row],h[col],e) -> MLP(+LN) -> scatter ----------------

__global__ __launch_bounds__(256) void edge_block_kernel(
    const __bf16* __restrict__ h_bf, __bf16* __restrict__ e_bf,
    float* __restrict__ e_f32, const int* __restrict__ ei,
    const __bf16* __restrict__ W1p, const float* __restrict__ b1,
    const float* __restrict__ g, const float* __restrict__ be,
    const __bf16* __restrict__ W2p, const float* __restrict__ b2,
    float* __restrict__ agg) {
  __shared__ __bf16 lds_mid[8][16 * HDIM];
  const int lane = threadIdx.x & 31;
  const int w = threadIdx.x >> 5;
  const int col = lane & 15;
  const int khalf = lane >> 4;
  const int rowoff = khalf * 8;
  const long tbase = ((long)blockIdx.x * 8 + w) * 16;

  long ar = tbase + col;
  if (ar >= NEDGES) ar = NEDGES - 1;
  const int er = ei[ar];
  const int ec = ei[NEDGES + ar];

  v8f acc[8];
#pragma unroll
  for (int nt = 0; nt < 8; ++nt) acc[nt] = v8f_zero();

  // layer 1: K = 384 = 12 chunks (h[row] | h[col] | e)
#pragma unroll
  for (int c = 0; c < 12; ++c) {
    const __bf16* rp;
    int kc;
    if (c < 4)       { rp = h_bf + (size_t)er * HDIM; kc = c * 32; }
    else if (c < 8)  { rp = h_bf + (size_t)ec * HDIM; kc = (c - 4) * 32; }
    else             { rp = e_bf + (size_t)ar * HDIM; kc = (c - 8) * 32; }
    v16bf a = load_arow(rp, kc, khalf);
#pragma unroll
    for (int nt = 0; nt < 8; ++nt)
      acc[nt] = wmma_bf16(a, load_btile(W1p, c * 8 + nt, lane), acc[nt]);
  }

  // bias + ReLU
  float g_l[8], be_l[8];
#pragma unroll
  for (int nt = 0; nt < 8; ++nt) {
    float bb = b1[nt * 16 + col];
    g_l[nt] = g[nt * 16 + col];
    be_l[nt] = be[nt * 16 + col];
#pragma unroll
    for (int m = 0; m < 8; ++m) {
      float v = acc[nt][m] + bb;
      acc[nt][m] = v > 0.f ? v : 0.f;
    }
  }

  // LayerNorm per row (row values spread over 16 lanes x 8 tiles)
#pragma unroll
  for (int m = 0; m < 8; ++m) {
    float s = 0.f, s2 = 0.f;
#pragma unroll
    for (int nt = 0; nt < 8; ++nt) { float v = acc[nt][m]; s += v; s2 += v * v; }
    s = hsum16(s);
    s2 = hsum16(s2);
    float mean = s * (1.f / HDIM);
    float var = fmaxf(s2 * (1.f / HDIM) - mean * mean, 0.f);
    float rstd = rsqrtf(var + LNEPS);
#pragma unroll
    for (int nt = 0; nt < 8; ++nt)
      acc[nt][m] = (acc[nt][m] - mean) * rstd * g_l[nt] + be_l[nt];
  }

  // C-layout -> A-layout via LDS (bf16)
#pragma unroll
  for (int m = 0; m < 8; ++m)
#pragma unroll
    for (int nt = 0; nt < 8; ++nt)
      lds_mid[w][(rowoff + m) * HDIM + nt * 16 + col] = (__bf16)acc[nt][m];
  asm volatile("s_wait_dscnt 0" ::: "memory");

  // layer 2: K = 128
#pragma unroll
  for (int nt = 0; nt < 8; ++nt) acc[nt] = v8f_zero();
  const __bf16* midrow = &lds_mid[w][col * HDIM];
#pragma unroll
  for (int c = 0; c < 4; ++c) {
    v16bf a = load_arow(midrow, c * 32, khalf);
#pragma unroll
    for (int nt = 0; nt < 8; ++nt)
      acc[nt] = wmma_bf16(a, load_btile(W2p, c * 8 + nt, lane), acc[nt]);
  }

  // receiver nodes for the output rows this half-wave owns
  int recv[8];
#pragma unroll
  for (int m = 0; m < 8; ++m) {
    long edge = tbase + rowoff + m;
    recv[m] = (edge < NEDGES) ? ei[NEDGES + edge] : 0;
  }

  // bias, scatter-add to agg, residual update of e (f32 + bf16 shadow)
#pragma unroll
  for (int nt = 0; nt < 8; ++nt) {
    float bb = b2[nt * 16 + col];
#pragma unroll
    for (int m = 0; m < 8; ++m) {
      long edge = tbase + rowoff + m;
      if (edge < NEDGES) {
        float val = acc[nt][m] + bb;
        int cg = nt * 16 + col;
        atomicAdd(&agg[(size_t)recv[m] * HDIM + cg], val);
        size_t off = (size_t)edge * HDIM + cg;
        float ev = e_f32[off] + val;
        e_f32[off] = ev;
        e_bf[off] = (__bf16)ev;
      }
    }
  }
}

// ---------------- node block: concat(h, agg/deg) -> MLP(+LN) -> residual ----------------

__global__ __launch_bounds__(256) void node_block_kernel(
    __bf16* __restrict__ h_bf, float* __restrict__ h_f32,
    const float* __restrict__ agg, const float* __restrict__ invdeg,
    const __bf16* __restrict__ W1p, const float* __restrict__ b1,
    const float* __restrict__ g, const float* __restrict__ be,
    const __bf16* __restrict__ W2p, const float* __restrict__ b2) {
  __shared__ __bf16 lds_mid[8][16 * HDIM];
  const int lane = threadIdx.x & 31;
  const int w = threadIdx.x >> 5;
  const int col = lane & 15;
  const int khalf = lane >> 4;
  const int rowoff = khalf * 8;
  const long tbase = ((long)blockIdx.x * 8 + w) * 16;

  long ar = tbase + col;
  if (ar >= NNODES) ar = NNODES - 1;
  const float inv = invdeg[ar];

  v8f acc[8];
#pragma unroll
  for (int nt = 0; nt < 8; ++nt) acc[nt] = v8f_zero();

  // layer 1: K = 256 = 8 chunks (h | agg/deg)
#pragma unroll
  for (int c = 0; c < 8; ++c) {
    v16bf a;
    if (c < 4)
      a = load_arow(h_bf + (size_t)ar * HDIM, c * 32, khalf);
    else
      a = build_a_f32scale(agg + (size_t)ar * HDIM, (c - 4) * 32, inv, khalf);
#pragma unroll
    for (int nt = 0; nt < 8; ++nt)
      acc[nt] = wmma_bf16(a, load_btile(W1p, c * 8 + nt, lane), acc[nt]);
  }

  float g_l[8], be_l[8];
#pragma unroll
  for (int nt = 0; nt < 8; ++nt) {
    float bb = b1[nt * 16 + col];
    g_l[nt] = g[nt * 16 + col];
    be_l[nt] = be[nt * 16 + col];
#pragma unroll
    for (int m = 0; m < 8; ++m) {
      float v = acc[nt][m] + bb;
      acc[nt][m] = v > 0.f ? v : 0.f;
    }
  }

#pragma unroll
  for (int m = 0; m < 8; ++m) {
    float s = 0.f, s2 = 0.f;
#pragma unroll
    for (int nt = 0; nt < 8; ++nt) { float v = acc[nt][m]; s += v; s2 += v * v; }
    s = hsum16(s);
    s2 = hsum16(s2);
    float mean = s * (1.f / HDIM);
    float var = fmaxf(s2 * (1.f / HDIM) - mean * mean, 0.f);
    float rstd = rsqrtf(var + LNEPS);
#pragma unroll
    for (int nt = 0; nt < 8; ++nt)
      acc[nt][m] = (acc[nt][m] - mean) * rstd * g_l[nt] + be_l[nt];
  }

#pragma unroll
  for (int m = 0; m < 8; ++m)
#pragma unroll
    for (int nt = 0; nt < 8; ++nt)
      lds_mid[w][(rowoff + m) * HDIM + nt * 16 + col] = (__bf16)acc[nt][m];
  asm volatile("s_wait_dscnt 0" ::: "memory");

#pragma unroll
  for (int nt = 0; nt < 8; ++nt) acc[nt] = v8f_zero();
  const __bf16* midrow = &lds_mid[w][col * HDIM];
#pragma unroll
  for (int c = 0; c < 4; ++c) {
    v16bf a = load_arow(midrow, c * 32, khalf);
#pragma unroll
    for (int nt = 0; nt < 8; ++nt)
      acc[nt] = wmma_bf16(a, load_btile(W2p, c * 8 + nt, lane), acc[nt]);
  }

#pragma unroll
  for (int nt = 0; nt < 8; ++nt) {
    float bb = b2[nt * 16 + col];
#pragma unroll
    for (int m = 0; m < 8; ++m) {
      long node = tbase + rowoff + m;
      if (node < NNODES) {
        float val = acc[nt][m] + bb;
        size_t off = (size_t)node * HDIM + nt * 16 + col;
        float hv = h_f32[off] + val;
        h_f32[off] = hv;
        h_bf[off] = (__bf16)hv;
      }
    }
  }
}

// ---------------- decoder: Linear(128->128)+ReLU+Linear(128->3) ----------------

__global__ __launch_bounds__(256) void decoder_kernel(
    const __bf16* __restrict__ h_bf, const __bf16* __restrict__ W1p,
    const float* __restrict__ b1, const __bf16* __restrict__ W2p,
    const float* __restrict__ b2, float* __restrict__ out) {
  __shared__ __bf16 lds_mid[8][16 * HDIM];
  const int lane = threadIdx.x & 31;
  const int w = threadIdx.x >> 5;
  const int col = lane & 15;
  const int khalf = lane >> 4;
  const int rowoff = khalf * 8;
  const long tbase = ((long)blockIdx.x * 8 + w) * 16;

  long ar = tbase + col;
  if (ar >= NNODES) ar = NNODES - 1;

  v8f acc[8];
#pragma unroll
  for (int nt = 0; nt < 8; ++nt) acc[nt] = v8f_zero();
#pragma unroll
  for (int c = 0; c < 4; ++c) {
    v16bf a = load_arow(h_bf + (size_t)ar * HDIM, c * 32, khalf);
#pragma unroll
    for (int nt = 0; nt < 8; ++nt)
      acc[nt] = wmma_bf16(a, load_btile(W1p, c * 8 + nt, lane), acc[nt]);
  }
#pragma unroll
  for (int nt = 0; nt < 8; ++nt) {
    float bb = b1[nt * 16 + col];
#pragma unroll
    for (int m = 0; m < 8; ++m) {
      float v = acc[nt][m] + bb;
      acc[nt][m] = v > 0.f ? v : 0.f;
    }
  }
#pragma unroll
  for (int m = 0; m < 8; ++m)
#pragma unroll
    for (int nt = 0; nt < 8; ++nt)
      lds_mid[w][(rowoff + m) * HDIM + nt * 16 + col] = (__bf16)acc[nt][m];
  asm volatile("s_wait_dscnt 0" ::: "memory");

  v8f acc2 = v8f_zero();
  const __bf16* midrow = &lds_mid[w][col * HDIM];
#pragma unroll
  for (int c = 0; c < 4; ++c) {
    v16bf a = load_arow(midrow, c * 32, khalf);
    acc2 = wmma_bf16(a, load_btile(W2p, c, lane), acc2);
  }
#pragma unroll
  for (int m = 0; m < 8; ++m) {
    long node = tbase + rowoff + m;
    if (node < NNODES && col < 3) out[node * 3 + col] = acc2[m] + b2[col];
  }
}

// ---------------- host launcher ----------------

extern "C" void kernel_launch(void* const* d_in, const int* in_sizes, int n_in,
                              void* d_out, int out_size, void* d_ws,
                              size_t ws_size, hipStream_t stream) {
  (void)in_sizes; (void)out_size; (void)ws_size;
  const float* x  = (const float*)d_in[0];
  const int*   ei = (const int*)d_in[1];
  const float* ea = (const float*)d_in[2];
  const float* P[96];
  for (int i = 3; i < n_in && i < 96; ++i) P[i] = (const float*)d_in[i];

  char* ws = (char*)d_ws;
  auto carve = [&](size_t bytes) -> void* {
    void* p = (void*)ws;
    ws += (bytes + 255) & ~(size_t)255;
    return p;
  };
  float*  h_f32 = (float*) carve((size_t)NNODES * HDIM * 4);
  __bf16* h_bf  = (__bf16*)carve((size_t)NNODES * HDIM * 2);
  float*  e_f32 = (float*) carve((size_t)NEDGES * HDIM * 4);
  __bf16* e_bf  = (__bf16*)carve((size_t)NEDGES * HDIM * 2);
  float*  agg   = (float*) carve((size_t)NNODES * HDIM * 4);
  float*  invd  = (float*) carve((size_t)NNODES * 4);

  auto pack = [&](const float* W, int Kact, int Nact, int KT, int NTn) -> __bf16* {
    int total = KT * NTn * 512;
    __bf16* out = (__bf16*)carve((size_t)total * 2);
    repack_w<<<(total + 255) / 256, 256, 0, stream>>>(W, out, Kact, Nact, NTn, total);
    return out;
  };

  // params flattened depth-first in setup_inputs() insertion order
  __bf16* neW1 = pack(P[3], 12, HDIM, 1, 8);     const float* neB1 = P[4];
  __bf16* neW2 = pack(P[5], HDIM, HDIM, 4, 8);   const float* neB2 = P[6];
  __bf16* eeW1 = pack(P[7], 4, HDIM, 1, 8);      const float* eeB1 = P[8];
  __bf16* eeW2 = pack(P[9], HDIM, HDIM, 4, 8);   const float* eeB2 = P[10];
  __bf16* deW1 = pack(P[11], HDIM, HDIM, 4, 8);  const float* deB1 = P[12];
  __bf16* deW2 = pack(P[13], HDIM, 3, 4, 1);     const float* deB2 = P[14];

  __bf16 *beW1[6], *beW2[6], *bnW1[6], *bnW2[6];
  const float *beB1[6], *beB2[6], *beG[6], *beBe[6];
  const float *bnB1[6], *bnB2[6], *bnG[6], *bnBe[6];
  for (int i = 0; i < 6; ++i) {
    int b = 15 + 12 * i;
    beW1[i] = pack(P[b + 0], 3 * HDIM, HDIM, 12, 8);
    beB1[i] = P[b + 1];
    beW2[i] = pack(P[b + 2], HDIM, HDIM, 4, 8);
    beB2[i] = P[b + 3];
    beG[i] = P[b + 4];
    beBe[i] = P[b + 5];
    bnW1[i] = pack(P[b + 6], 2 * HDIM, HDIM, 8, 8);
    bnB1[i] = P[b + 7];
    bnW2[i] = pack(P[b + 8], HDIM, HDIM, 4, 8);
    bnB2[i] = P[b + 9];
    bnG[i] = P[b + 10];
    bnBe[i] = P[b + 11];
  }

  // scatter-mean degrees
  zero_f32<<<(NNODES + 255) / 256, 256, 0, stream>>>(invd, NNODES);
  deg_kernel<<<(NEDGES + 255) / 256, 256, 0, stream>>>(ei, invd);
  invdeg_kernel<<<(NNODES + 255) / 256, 256, 0, stream>>>(invd);

  const int nodeBlocks = (((NNODES + 15) / 16) + 7) / 8;  // 391
  const int edgeBlocks = (((NEDGES + 15) / 16) + 7) / 8;  // 6250

  encoder_kernel<<<nodeBlocks, 256, 0, stream>>>(x, 12, (long)NNODES, neW1, neB1,
                                                 neW2, neB2, h_f32, h_bf);
  encoder_kernel<<<edgeBlocks, 256, 0, stream>>>(ea, 4, (long)NEDGES, eeW1, eeB1,
                                                 eeW2, eeB2, e_f32, e_bf);

  for (int i = 0; i < 6; ++i) {
    zero_f32<<<((long)NNODES * HDIM + 255) / 256, 256, 0, stream>>>(
        agg, (long)NNODES * HDIM);
    edge_block_kernel<<<edgeBlocks, 256, 0, stream>>>(
        h_bf, e_bf, e_f32, ei, beW1[i], beB1[i], beG[i], beBe[i], beW2[i],
        beB2[i], agg);
    node_block_kernel<<<nodeBlocks, 256, 0, stream>>>(
        h_bf, h_f32, agg, invd, bnW1[i], bnB1[i], bnG[i], bnBe[i], bnW2[i],
        bnB2[i]);
  }

  decoder_kernel<<<nodeBlocks, 256, 0, stream>>>(h_bf, deW1, deB1, deW2, deB2,
                                                 (float*)d_out);
}
